// RNN_74328704025084
// MI455X (gfx1250) — matrix operand order repaired and checked by reading, not compiled, not run on hardware
//
#include <hip/hip_runtime.h>
#include <hip/hip_bf16.h>

typedef __attribute__((ext_vector_type(16))) __bf16 v16bf;
typedef __attribute__((ext_vector_type(8)))  __bf16 v8bf;
typedef __attribute__((ext_vector_type(8)))  float  v8f;
typedef __attribute__((ext_vector_type(4)))  float  v4f;
typedef __attribute__((ext_vector_type(4)))  unsigned int v4u;
typedef __attribute__((ext_vector_type(8)))  int v8i;
typedef __attribute__((ext_vector_type(4)))  int v4i;

#define T_STEPS 512
#define BATCH   128
#define NIN     256
#define NH      512

__device__ __forceinline__ v16bf cat16(v8bf lo, v8bf hi) {
  v16bf r;
#pragma unroll
  for (int j = 0; j < 8; ++j) { r[j] = lo[j]; r[8 + j] = hi[j]; }
  return r;
}

// tanh(x) = (e^2x - 1)/(e^2x + 1); clamp so e^2x stays finite (x=15 already
// saturates bf16-level precision). ~6 VALU ops vs OCML's long sequence.
__device__ __forceinline__ float fast_tanh(float x) {
  x = fminf(fmaxf(x, -15.0f), 15.0f);
  float e = __expf(2.0f * x);
  return __fdividef(e - 1.0f, e + 1.0f);
}

// ---------------------------------------------------------------------------
// Kernel 0: W (f32 [512x768]) -> bf16 Wx [512x256], Wh [512x512] (row-major,
// K contiguous so WMMA B fragments are contiguous B128 loads); zero h0 slot.
// ---------------------------------------------------------------------------
__global__ void prep_kernel(const float* __restrict__ W, float* __restrict__ out,
                            __bf16* __restrict__ wxb, __bf16* __restrict__ whb) {
  int idx = blockIdx.x * blockDim.x + threadIdx.x;
  if (idx < NH * (NIN + NH)) {
    int n = idx / (NIN + NH);
    int c = idx - n * (NIN + NH);
    float w = W[idx];
    if (c < NIN) wxb[n * NIN + c] = (__bf16)w;
    else         whb[(size_t)n * NH + (c - NIN)] = (__bf16)w;
  }
  if (idx < BATCH * NH) out[idx] = 0.0f;   // h0 = zeros
}

// ---------------------------------------------------------------------------
// Kernel 1: xp = x @ Wx.T + b (M=65536, N=512, K=256). One wave: 16x128 strip,
// A (f32->bf16 on the fly) reused across 8 N tiles. Writes f32 into d_out
// slots 1..512 (the scan later overwrites in place).
// ---------------------------------------------------------------------------
__global__ __launch_bounds__(256) void xp_gemm_kernel(
    const float* __restrict__ x, const float* __restrict__ bias,
    const __bf16* __restrict__ wxb, float* __restrict__ out) {
  const int wid  = blockIdx.x * (blockDim.x >> 5) + (threadIdx.x >> 5);
  const int lane = threadIdx.x & 31;
  const int lm = lane & 15, lk = lane >> 4;
  const int mtile = wid & 4095;
  const int ng    = wid >> 12;
  const int m0 = mtile * 16;
  const int n0 = ng * 128;

  v8f acc[8];
#pragma unroll
  for (int nt = 0; nt < 8; ++nt) {
    float bv = bias[n0 + nt * 16 + lm];
#pragma unroll
    for (int v = 0; v < 8; ++v) acc[nt][v] = bv;
  }

  const float* arow = x + (size_t)(m0 + lm) * NIN;
  const __bf16* bptr[8];
#pragma unroll
  for (int nt = 0; nt < 8; ++nt)
    bptr[nt] = wxb + (size_t)(n0 + nt * 16 + lm) * NIN + lk * 16;

#pragma unroll
  for (int ks = 0; ks < 8; ++ks) {
    const int kb = ks * 32;
    v4f c0 = *(const v4f*)(arow + kb + lk * 16);
    v4f c1 = *(const v4f*)(arow + kb + lk * 16 + 4);
    v4f c2 = *(const v4f*)(arow + kb + lk * 16 + 8);
    v4f c3 = *(const v4f*)(arow + kb + lk * 16 + 12);
    v16bf a;
#pragma unroll
    for (int j = 0; j < 4; ++j) {
      a[j]      = (__bf16)c0[j];
      a[4 + j]  = (__bf16)c1[j];
      a[8 + j]  = (__bf16)c2[j];
      a[12 + j] = (__bf16)c3[j];
    }
#pragma unroll
    for (int nt = 0; nt < 8; ++nt) {
      v16bf bf = cat16(*(const v8bf*)(bptr[nt] + kb),
                       *(const v8bf*)(bptr[nt] + kb + 8));
      acc[nt] = __builtin_amdgcn_wmma_f32_16x16x32_bf16(
          false, a, false, bf, (short)0, acc[nt], false, false);
    }
  }

  float* obase = out + (size_t)BATCH * NH;
#pragma unroll
  for (int nt = 0; nt < 8; ++nt) {
    const int n = n0 + nt * 16 + lm;
#pragma unroll
    for (int v = 0; v < 8; ++v)
      obase[(size_t)(m0 + v + lk * 8) * NH + n] = acc[nt][v];
  }
}

// ---------------------------------------------------------------------------
// Kernel 2: persistent recurrent scan. 8 WGs x 512 threads (16 waves); WG owns
// 16 batch rows (recurrence independent across batch -> no global sync, one
// s_barrier/step). h double-buffered in LDS (bf16, padded stride 520); Wh K<256
// half TDM-loaded into LDS once (320KB/WGP), K>=256 half stays L2-resident.
// xp for step t+1 is prefetched into registers during step t's WMMAs.
// ---------------------------------------------------------------------------
__global__ __launch_bounds__(512, 1) void rnn_scan_kernel(
    const __bf16* __restrict__ whb, float* __restrict__ out) {
  __shared__ __attribute__((aligned(16))) __bf16 whLds[NH][264];      // 270,336 B
  __shared__ __attribute__((aligned(16))) __bf16 hbuf[2][16][520];    //  33,280 B

  const int tid  = threadIdx.x;
  const int wave = tid >> 5, lane = tid & 31;
  const int lm = lane & 15, lk = lane >> 4;
  const int b0 = blockIdx.x * 16;
  const int n0 = wave * 32;

#if __has_builtin(__builtin_amdgcn_tensor_load_to_lds)
  // TDM: DMA Wh[:, 0:256) (2D tile 256x512 of 2B elems from a 512-stride
  // tensor) into whLds. pad_interval=128 dwords (code 6) + pad_amount=4 dwords
  // (code 3) reproduces the 264-element padded row stride. Wave 0 issues,
  // waits TENSORcnt, then the workgroup barrier publishes the data.
  if (wave == 0) {
    unsigned lds_base = (unsigned)(size_t)(&whLds[0][0]);
    unsigned long long ga = (unsigned long long)(size_t)whb;
    v4u g0;
    g0[0] = 1u;                                        // count=1 (valid D#)
    g0[1] = lds_base;                                  // lds_addr
    g0[2] = (unsigned)(ga & 0xFFFFFFFFu);              // global_addr[31:0]
    g0[3] = (unsigned)((ga >> 32) & 0x01FFFFFFu)       // global_addr[56:32]
          | 0x80000000u;                               // type=2 (image)
    v8i g1;
    g1[0] = (int)(0x10000u            // data_size=1 -> 2 bytes
                | 0x100000u           // pad_enable
                | (6u << 22)          // pad_interval: 128 dwords
                | (3u << 25));        // pad_amount: 4 dwords
    g1[1] = (int)(512u << 16);        // tensor_dim0 = 512 (bits 79:48, low16)
    g1[2] = (int)(512u << 16);        // tensor_dim1 = 512 (bits 111:80, low16)
    g1[3] = (int)(256u << 16);        // tile_dim0 = 256 (bits 127:112)
    g1[4] = (int)512;                 // tile_dim1 = 512 (bits 143:128)
    g1[5] = (int)512;                 // tensor_dim0_stride = 512
    g1[6] = 0;
    g1[7] = 0;
    v4i gz; gz[0] = 0; gz[1] = 0; gz[2] = 0; gz[3] = 0;
    v8i gz8;
#pragma unroll
    for (int j = 0; j < 8; ++j) gz8[j] = 0;
    // 6-arg form (clang-23 / therock-10.0 headers): groups 2/3 unused (2D).
    __builtin_amdgcn_tensor_load_to_lds(g0, g1, gz, gz, gz8, 0);
    __builtin_amdgcn_s_wait_tensorcnt(0);
  }
#else
  for (int c = tid; c < NH * 32; c += 512) {
    int n  = c >> 5;
    int k8 = (c & 31) * 8;
    *(v8bf*)&whLds[n][k8] = *(const v8bf*)(whb + (size_t)n * NH + k8);
  }
#endif
  // h0 = 0
  {
    uint4 z; z.x = 0; z.y = 0; z.z = 0; z.w = 0;
    for (int c = tid; c < (16 * 520 * 2) / 16; c += 512)
      ((uint4*)&hbuf[0][0][0])[c] = z;
  }

  // Per-wave invariant B pointers (hoisted out of the time loop).
  const int nA = n0 + lm, nB = n0 + 16 + lm;
  const __bf16* gbpA = whb + (size_t)nA * NH + lk * 16;   // K>=256 half (L2)
  const __bf16* gbpB = whb + (size_t)nB * NH + lk * 16;
  const __bf16* lbpA = &whLds[nA][lk * 16];               // K<256 half (LDS)
  const __bf16* lbpB = &whLds[nB][lk * 16];
  const size_t rowA = (size_t)(b0 + lk * 8) * NH;         // xp/out row base

  // Prologue: load xp for t=0 (slot 1).
  v8f xpn[2];
  {
    const float* p0 = out + (size_t)1 * BATCH * NH + rowA;
#pragma unroll
    for (int nt = 0; nt < 2; ++nt) {
      const int n = n0 + nt * 16 + lm;
#pragma unroll
      for (int v = 0; v < 8; ++v) xpn[nt][v] = p0[(size_t)v * NH + n];
    }
  }
  __syncthreads();

#pragma unroll 1
  for (int t = 0; t < T_STEPS; ++t) {
    const int cur = t & 1, nxt = cur ^ 1;
    float* obase = out + (size_t)(t + 1) * BATCH * NH;

    v8f acc[2];
    acc[0] = xpn[0];
    acc[1] = xpn[1];

    // Prefetch xp for step t+1 (slot t+2; clamped, unused value at t=511).
    {
      const int tn = (t < T_STEPS - 1) ? (t + 2) : (t + 1);
      const float* pn = out + (size_t)tn * BATCH * NH + rowA;
#pragma unroll
      for (int nt = 0; nt < 2; ++nt) {
        const int n = n0 + nt * 16 + lm;
#pragma unroll
        for (int v = 0; v < 8; ++v) xpn[nt][v] = pn[(size_t)v * NH + n];
      }
    }

#pragma unroll
    for (int ks = 0; ks < 16; ++ks) {
      const int kb = ks * 32;
      v16bf a = cat16(*(const v8bf*)&hbuf[cur][lm][kb + lk * 16],
                      *(const v8bf*)&hbuf[cur][lm][kb + lk * 16 + 8]);
      v8bf b0a, b0b, b1a, b1b;
      if (kb < 256) {                      // compile-time after unroll
        b0a = *(const v8bf*)(lbpA + kb);  b0b = *(const v8bf*)(lbpA + kb + 8);
        b1a = *(const v8bf*)(lbpB + kb);  b1b = *(const v8bf*)(lbpB + kb + 8);
      } else {
        b0a = *(const v8bf*)(gbpA + kb);  b0b = *(const v8bf*)(gbpA + kb + 8);
        b1a = *(const v8bf*)(gbpB + kb);  b1b = *(const v8bf*)(gbpB + kb + 8);
      }
      acc[0] = __builtin_amdgcn_wmma_f32_16x16x32_bf16(
          false, a, false, cat16(b0a, b0b), (short)0, acc[0], false, false);
      acc[1] = __builtin_amdgcn_wmma_f32_16x16x32_bf16(
          false, a, false, cat16(b1a, b1b), (short)0, acc[1], false, false);
    }

    // h = tanh(acc): f32 to output (in place over xp), bf16 to LDS for t+1.
#pragma unroll
    for (int nt = 0; nt < 2; ++nt) {
      const int n = n0 + nt * 16 + lm;
#pragma unroll
      for (int v = 0; v < 8; ++v) {
        float hv = fast_tanh(acc[nt][v]);
        obase[rowA + (size_t)v * NH + n] = hv;
        hbuf[nxt][v + lk * 8][n] = (__bf16)hv;
      }
    }
    __syncthreads();
  }
}

// ---------------------------------------------------------------------------
extern "C" void kernel_launch(void* const* d_in, const int* in_sizes, int n_in,
                              void* d_out, int out_size, void* d_ws, size_t ws_size,
                              hipStream_t stream) {
  const float* x = (const float*)d_in[0];   // (512,128,256) f32
  const float* W = (const float*)d_in[1];   // (512,768) f32
  const float* b = (const float*)d_in[2];   // (512,) f32
  float* out = (float*)d_out;               // (513,128,512) f32

  __bf16* wxb = (__bf16*)d_ws;              // 512*256 bf16
  __bf16* whb = wxb + NH * NIN;             // 512*512 bf16

  prep_kernel<<<1536, 256, 0, stream>>>(W, out, wxb, whb);
  xp_gemm_kernel<<<2048, 256, 0, stream>>>(x, b, wxb, out);
  rnn_scan_kernel<<<8, 512, 0, stream>>>(whb, out);
}
